// ScaledDotProductAttention_29996051595373
// MI455X (gfx1250) — matrix operand order repaired and checked by reading, not compiled, not run on hardware
//
#include <hip/hip_runtime.h>

typedef _Float16 v16h __attribute__((ext_vector_type(16)));
typedef _Float16 v8h  __attribute__((ext_vector_type(8)));
typedef float    v8f  __attribute__((ext_vector_type(8)));
typedef float    v4f  __attribute__((ext_vector_type(4)));
typedef int      v4i  __attribute__((ext_vector_type(4)));

#define B_ 2
#define H_ 4
#define S_ 4096
#define D_ 64
#define WAVES 8   // 8 wave32s per block, each wave owns 16 s-rows
#define VTS 40    // padded t-stride for transposed V in LDS (keeps 16B align)

__device__ __forceinline__ v8f wmma_f16(v16h a, v16h b, v8f c) {
  // D = A(16x32 f16) * B(32x16 f16) + C(16x16 f32)
  return __builtin_amdgcn_wmma_f32_16x16x32_f16(false, a, false, b,
                                                (short)0, c, false, false);
}
__device__ __forceinline__ v16h cat8(v8h lo, v8h hi) {
  return __builtin_shufflevector(lo, hi, 0, 1, 2, 3, 4, 5, 6, 7,
                                 8, 9, 10, 11, 12, 13, 14, 15);
}

__global__ __launch_bounds__(32 * WAVES)
void attn_batch_softmax_fused(const float* __restrict__ q,
                              const float* __restrict__ k,
                              const float* __restrict__ v,
                              const int*   __restrict__ mask,
                              float* __restrict__ out,
                              float* __restrict__ attn)
{
  // double-buffered f16 tiles shared by all 8 waves (K row-major, V transposed)
  __shared__ __align__(16) _Float16 ldsK [2][B_][32][D_];
  __shared__ __align__(16) _Float16 ldsVT[2][B_][D_][VTS];

  const int tid   = threadIdx.x;
  const int lane  = tid & 31;
  const int wave  = tid >> 5;
  const int lmod  = lane & 15;   // N index within tiles / A-row index
  const int lhalf = lane >> 4;   // which half-wave
  const int h     = blockIdx.y;
  const int s0    = (blockIdx.x * WAVES + wave) * 16;

  const v8f vzero = {};

  // cooperative staging role: each thread owns 16 floats of K and 16 of V
  const int cb   = tid >> 7;         // batch
  const int crow = (tid >> 2) & 31;  // t-row within 32-row tile
  const int cseg = tid & 3;          // 16-float d-segment
  const float* kg_base = k + ((size_t)(cb * H_ + h) * S_ + crow) * D_ + cseg * 16;
  const float* vg_base = v + ((size_t)(cb * H_ + h) * S_ + crow) * D_ + cseg * 16;

  auto stage_regs = [&](int tb, v4f* kr, v4f* vr) {
    const v4f* kp = (const v4f*)(kg_base + (size_t)tb * D_);
    const v4f* vp = (const v4f*)(vg_base + (size_t)tb * D_);
    #pragma unroll
    for (int g = 0; g < 4; ++g) { kr[g] = kp[g]; vr[g] = vp[g]; }
  };
  auto stage_lds = [&](int buf, const v4f* kr, const v4f* vr) {
    v8h k0, k1;
    #pragma unroll
    for (int e = 0; e < 4; ++e) {
      k0[e]     = (_Float16)kr[0][e];
      k0[4 + e] = (_Float16)kr[1][e];
      k1[e]     = (_Float16)kr[2][e];
      k1[4 + e] = (_Float16)kr[3][e];
    }
    *(v8h*)&ldsK[buf][cb][crow][cseg * 16]     = k0;   // ds_store_b128
    *(v8h*)&ldsK[buf][cb][crow][cseg * 16 + 8] = k1;   // ds_store_b128
    #pragma unroll
    for (int j = 0; j < 16; ++j)                       // transpose V into LDS
      ldsVT[buf][cb][cseg * 16 + j][crow] = (_Float16)vr[j >> 2][j & 3];
  };

  // ---------------- Q^T B-fragments (K-dim = d), loaded once ----------------
  // B-frag (32x16 f16): lane holds col n = s0+lmod, K = d = c*32 + 16*lhalf + i
  v16h qf[2][2];
  #pragma unroll
  for (int b = 0; b < 2; ++b) {
    const float* qrow = q + ((size_t)(b * H_ + h) * S_ + (s0 + lmod)) * D_;
    #pragma unroll
    for (int c = 0; c < 2; ++c) {
      const v4f* p4 = (const v4f*)(qrow + c * 32 + 16 * lhalf);
      #pragma unroll
      for (int g = 0; g < 4; ++g) {
        v4f x = p4[g];
        qf[b][c][4 * g + 0] = (_Float16)x[0];
        qf[b][c][4 * g + 1] = (_Float16)x[1];
        qf[b][c][4 * g + 2] = (_Float16)x[2];
        qf[b][c][4 * g + 3] = (_Float16)x[3];
      }
    }
  }

  // ---------------- output accumulators: O[b][d-tile] (16x16 f32) -----------
  v8f O[2][4];
  #pragma unroll
  for (int b = 0; b < 2; ++b)
    #pragma unroll
    for (int dt = 0; dt < 4; ++dt)
      O[b][dt] = vzero;

  const int* mrowbase = mask + (size_t)(s0 + lmod) * S_;

  // prologue: stage first tile into buffer 0
  v4f kr[4], vr[4];
  stage_regs(0, kr, vr);
  stage_lds(0, kr, vr);
  __syncthreads();

  for (int tb = 0; tb < S_; tb += 32) {
    const int cur = (tb >> 5) & 1;
    const int tbn = tb + 32;
    if (tbn < S_) stage_regs(tbn, kr, vr);  // prefetch next tile into regs

    // ---- transposed scores from LDS K: S^T tile, M = t, N = s --------------
    v8f sc[2][2];
    #pragma unroll
    for (int b = 0; b < 2; ++b) {
      #pragma unroll
      for (int sub = 0; sub < 2; ++sub) {
        v8f acc = vzero;
        #pragma unroll
        for (int c = 0; c < 2; ++c) {
          // A-frag halves 0..7: d = c*32 + 16*lhalf + i ; halves 8..15: +8
          const v8h* plo =
              (const v8h*)&ldsK[cur][b][sub * 16 + lmod][c * 32 + 16 * lhalf];
          const v8h* phi =
              (const v8h*)&ldsK[cur][b][sub * 16 + lmod][c * 32 + 16 * lhalf + 8];
          acc = wmma_f16(cat8(plo[0], phi[0]), qf[b][c], acc);
        }
        sc[b][sub] = acc;
      }
    }

    // ---- mask + pairwise (batch-axis) softmax + non-temporal attn stores ----
    // lane holds s = s0+lmod fixed; reg r of subtile 'sub' is
    // t = tb + sub*16 + 8*lhalf + r
    #pragma unroll
    for (int sub = 0; sub < 2; ++sub) {
      const int toff = sub * 16 + 8 * lhalf;
      v4i ml = *(const v4i*)(mrowbase + tb + toff);
      v4i mh = *(const v4i*)(mrowbase + tb + toff + 4);
      const int mv[8] = {ml[0], ml[1], ml[2], ml[3], mh[0], mh[1], mh[2], mh[3]};
      #pragma unroll
      for (int r = 0; r < 8; ++r) {
        float a0   = sc[0][sub][r];
        float a1   = sc[1][sub][r];
        // masked: both entries forced to NEG -> diff 0 -> p = 0.5 (matches ref)
        float diff = mv[r] ? 0.0f : (a0 - a1) * 0.125f;  // 1/sqrt(64)
        float p    = 1.0f / (1.0f + __expf(-diff));      // softmax over B=2
        sc[0][sub][r] = p;
        sc[1][sub][r] = 1.0f - p;
      }
      #pragma unroll
      for (int b = 0; b < 2; ++b) {
        float* arow = attn + ((size_t)(b * H_ + h) * S_ + (s0 + lmod)) * S_ +
                      tb + toff;
        v4f lo = __builtin_shufflevector(sc[b][sub], sc[b][sub], 0, 1, 2, 3);
        v4f hi = __builtin_shufflevector(sc[b][sub], sc[b][sub], 4, 5, 6, 7);
        __builtin_nontemporal_store(lo, (v4f*)arow);        // 537 MB stream:
        __builtin_nontemporal_store(hi, (v4f*)(arow + 4));  // bypass caches
      }
    }

    // ---- P as f16 A-fragment: pure per-lane repack (layouts coincide) ------
    v16h pf[2];
    #pragma unroll
    for (int b = 0; b < 2; ++b)
      #pragma unroll
      for (int i = 0; i < 8; ++i) {
        pf[b][i]     = (_Float16)sc[b][0][i];  // t = tb +      8*lhalf + i
        pf[b][8 + i] = (_Float16)sc[b][1][i];  // t = tb + 16 + 8*lhalf + i
      }

    // ---- O += P * V  from transposed LDS V: two ds_load_b128 per frag ------
    #pragma unroll
    for (int b = 0; b < 2; ++b) {
      #pragma unroll
      for (int dt = 0; dt < 4; ++dt) {
        const v8h* pv =
            (const v8h*)&ldsVT[cur][b][dt * 16 + lmod][8 * lhalf];
        const v8h* pw =
            (const v8h*)&ldsVT[cur][b][dt * 16 + lmod][16 + 8 * lhalf];
        O[b][dt] = wmma_f16(pf[b], cat8(pv[0], pw[0]), O[b][dt]);
      }
    }

    if (tbn < S_) stage_lds(cur ^ 1, kr, vr);  // fill the other buffer
    __syncthreads();                            // one barrier per iteration
  }

  // ---------------- write out[b,h,s,d]: M = s-rel = r + 8*lhalf -------------
  #pragma unroll
  for (int b = 0; b < 2; ++b)
    #pragma unroll
    for (int dt = 0; dt < 4; ++dt)
      #pragma unroll
      for (int r = 0; r < 8; ++r)
        out[((size_t)(b * H_ + h) * S_ + s0 + r + 8 * lhalf) * D_ +
            dt * 16 + lmod] = O[b][dt][r];
}

extern "C" void kernel_launch(void* const* d_in, const int* in_sizes, int n_in,
                              void* d_out, int out_size, void* d_ws,
                              size_t ws_size, hipStream_t stream) {
  const float* q    = (const float*)d_in[0];
  const float* k    = (const float*)d_in[1];
  const float* v    = (const float*)d_in[2];
  const int*   mask = (const int*)d_in[3];
  float* out  = (float*)d_out;
  float* attn = out + (size_t)B_ * H_ * S_ * D_;  // tuple: (out, attn) flat

  dim3 grid(S_ / (WAVES * 16), H_);  // (32, 4)
  dim3 block(32 * WAVES);            // 256 threads = 8 wave32s
  attn_batch_softmax_fused<<<grid, block, 0, stream>>>(q, k, v, mask, out, attn);
}
